// MGDPR_3874060501527
// MI455X (gfx1250) — matrix operand order repaired and checked by reading, not compiled, not run on hardware
//
#include <hip/hip_runtime.h>
#include <hip/hip_bf16.h>

#define B_ 2
#define N_ 1500
#define T_ 20
#define DIN_ 32
#define C_ 128
#define R_ 5
#define K_ 5
#define L_ 3
#define OUT_ 2
#define EPS_ 1e-5f
#define M_ (B_ * N_)  // 3000 rows for node-level GEMMs

typedef __attribute__((ext_vector_type(16))) __bf16 v16bf;
typedef __attribute__((ext_vector_type(8)))  float  v8f;

union V16u {
  v16bf v;
  uint4 u[2];
  uint2 u2[4];
  __bf16 h[16];
};

__device__ __forceinline__ v8f wmma_bf16(v16bf a, v16bf b, v8f c) {
  return __builtin_amdgcn_wmma_f32_16x16x32_bf16(false, a, false, b, (short)0, c,
                                                 false, false);
}

__device__ __forceinline__ float waveSum(float x) {
#pragma unroll
  for (int off = 16; off > 0; off >>= 1) x += __shfl_xor(x, off, 32);
  return x;
}

__device__ __forceinline__ void cvt8(V16u& d, int base, float4 a, float4 b) {
  d.h[base + 0] = (__bf16)a.x; d.h[base + 1] = (__bf16)a.y;
  d.h[base + 2] = (__bf16)a.z; d.h[base + 3] = (__bf16)a.w;
  d.h[base + 4] = (__bf16)b.x; d.h[base + 5] = (__bf16)b.y;
  d.h[base + 6] = (__bf16)b.z; d.h[base + 7] = (__bf16)b.w;
}

// ---------------------------------------------------------------- utilities
__global__ void zero_kernel(float* p, int n) {
  int i = blockIdx.x * blockDim.x + threadIdx.x;
  if (i < n) p[i] = 0.f;
}

// adj fp32 -> bf16 once; all 3 layers then read half the bytes (L2-resident).
__global__ void cvt_adj_kernel(const float* __restrict__ adj,
                               __bf16* __restrict__ abf) {
  size_t i = ((size_t)blockIdx.x * 256 + threadIdx.x) * 4;
  const size_t total = (size_t)B_ * R_ * N_ * N_;  // 22,500,000 (mult of 4)
  if (i >= total) return;
  float4 f = *(const float4*)(adj + i);
  union { __bf16 h[4]; uint2 u; } p;
  p.h[0] = (__bf16)f.x; p.h[1] = (__bf16)f.y;
  p.h[2] = (__bf16)f.z; p.h[3] = (__bf16)f.w;
  *(uint2*)(abf + i) = p.u;
}

__global__ void embed_kernel(const float* __restrict__ x,
                             const float* __restrict__ emb_w,
                             const float* __restrict__ emb_b,
                             float* __restrict__ h) {
  int idx = blockIdx.x * blockDim.x + threadIdx.x;
  if (idx >= M_ * C_) return;
  int c = idx & (C_ - 1);
  int bn = idx >> 7;
  const float* xp = x + ((size_t)bn * T_ + (T_ - 1)) * DIN_;
  float s = emb_b[c];
#pragma unroll
  for (int d = 0; d < DIN_; ++d) s += xp[d] * emb_w[c * DIN_ + d];
  h[idx] = s;
}

__global__ void asum_kernel(const float* __restrict__ alpha, int l,
                            float* __restrict__ asum) {
  int r = threadIdx.x;
  if (r >= R_) return;
  const float* a = alpha + ((size_t)l * R_ + r) * K_;
  float mx = a[0];
#pragma unroll
  for (int k = 1; k < K_; ++k) mx = fmaxf(mx, a[k]);
  float se = 0.f;
#pragma unroll
  for (int k = 0; k < K_; ++k) se += __expf(a[k] - mx);
  float s = 0.f;
#pragma unroll
  for (int k = 0; k < K_; ++k) s += __expf(a[k] - mx) / se;
  asum[r] = s;
}

// Wms[r,o,c] = conv_w[l,r] * mean_k transition[l,r,k,o,c]
__global__ void prep_kernel(const float* __restrict__ transition,
                            const float* __restrict__ conv_w, int l,
                            float* __restrict__ Wms) {
  int idx = blockIdx.x * blockDim.x + threadIdx.x;
  if (idx >= R_ * C_ * C_) return;
  int r = idx / (C_ * C_);
  int oc = idx - r * C_ * C_;
  const float* t0 = transition + (((size_t)l * R_ + r) * K_) * C_ * C_ + oc;
  float s = 0.f;
#pragma unroll
  for (int k = 0; k < K_; ++k) s += t0[(size_t)k * C_ * C_];
  Wms[idx] = s * (1.f / K_) * conv_w[l * R_ + r];
}

// ------------------- P[b,r] = asum[r] * (adj_bf16[b,r] @ h[b]);  N x N x C
// 4 waves / 64 rows per block; each wave: 16 rows x all 8 col tiles of C=128.
__global__ __launch_bounds__(128) void diffusion_kernel(
    const __bf16* __restrict__ abf, const float* __restrict__ h,
    const float* __restrict__ asum, float* __restrict__ P) {
  int br = blockIdx.y;
  int b = br / R_, r = br - b * R_;
  int wave = threadIdx.x >> 5, lane = threadIdx.x & 31;
  int laneM = lane & 15, kh = lane >> 4;
  int mrow = blockIdx.x * 64 + wave * 16;
  int arow = mrow + laneM;
  arow = arow < N_ ? arow : N_ - 1;  // clamp: OOB rows only feed OOB stores

  __shared__ __align__(16) __bf16 hsm[128][40];  // [c][k], padded rows

  v8f acc[8];
#pragma unroll
  for (int t = 0; t < 8; ++t) acc[t] = (v8f){0, 0, 0, 0, 0, 0, 0, 0};

  const __bf16* abfMat = abf + ((size_t)(b * R_ + r)) * N_ * N_;
  const float* hsrc = h + (size_t)b * N_ * C_;
  int cc = threadIdx.x;  // 0..127

  for (int k0 = 0; k0 < N_; k0 += 32) {
    bool full = (k0 + 32) <= N_;  // uniform over block
    // stage h[k0:k0+32, :] transposed bf16 -> LDS (128 threads, 2 halves each)
#pragma unroll
    for (int half = 0; half < 2; ++half) {
      int kk = half * 16;
      V16u p;
      if (full) {
#pragma unroll
        for (int j = 0; j < 16; ++j)
          p.h[j] = (__bf16)hsrc[(size_t)(k0 + kk + j) * C_ + cc];
      } else {
#pragma unroll
        for (int j = 0; j < 16; ++j) {
          int k = k0 + kk + j;
          int kc = k < N_ ? k : N_ - 1;
          float v = hsrc[(size_t)kc * C_ + cc];
          p.h[j] = (__bf16)(k < N_ ? v : 0.f);
        }
      }
      uint4* dst = (uint4*)&hsm[cc][kk];
      dst[0] = p.u[0];
      dst[1] = p.u[1];
    }
    __syncthreads();

    // A tile 16x32 bf16: rows are 3000B (8B aligned) -> 4x global_load_b64
    V16u av;
    const __bf16* rowp = abfMat + (size_t)arow * N_ + k0;
    if (full) {
      const uint2* q0 = (const uint2*)(rowp + kh * 8);
      const uint2* q1 = (const uint2*)(rowp + 16 + kh * 8);
      av.u2[0] = q0[0];
      av.u2[1] = q0[1];
      av.u2[2] = q1[0];
      av.u2[3] = q1[1];
    } else {
      int valid = N_ - k0;
#pragma unroll
      for (int j = 0; j < 8; ++j) {
        int k = kh * 8 + j;
        __bf16 vv = rowp[k < valid ? k : 0];
        av.h[j] = k < valid ? vv : (__bf16)0.f;
      }
#pragma unroll
      for (int j = 0; j < 8; ++j) {
        int k = 16 + kh * 8 + j;
        __bf16 vv = rowp[k < valid ? k : 0];
        av.h[8 + j] = k < valid ? vv : (__bf16)0.f;
      }
    }

    // prefetch all 8 B tiles, then 8 wmmas
    V16u pb[8];
#pragma unroll
    for (int t = 0; t < 8; ++t) {
      const uint4* bp = (const uint4*)&hsm[t * 16 + laneM][kh * 8];
      const uint4* bq = (const uint4*)&hsm[t * 16 + laneM][16 + kh * 8];
      pb[t].u[0] = bp[0];
      pb[t].u[1] = bq[0];
    }
#pragma unroll
    for (int t = 0; t < 8; ++t) acc[t] = wmma_bf16(av.v, pb[t].v, acc[t]);
    __syncthreads();
  }

  float s = asum[r];
  float* Pb = P + ((size_t)(b * R_ + r)) * N_ * C_;
#pragma unroll
  for (int t = 0; t < 8; ++t) {
#pragma unroll
    for (int vv = 0; vv < 8; ++vv) {
      int m = mrow + vv + 8 * kh;
      if (m < N_) Pb[(size_t)m * C_ + t * 16 + laneM] = s * acc[t][vv];
    }
  }
}

// ---------------- hdiff = relu(sum_r P[b,r] @ Wms[r]^T + conv_b[l]);  K = R*C
__global__ __launch_bounds__(128) void mix_kernel(
    const float* __restrict__ P, const float* __restrict__ Wms,
    const float* __restrict__ conv_b, int l, float* __restrict__ hdiff) {
  int b = blockIdx.y;
  int wave = threadIdx.x >> 5, lane = threadIdx.x & 31;
  int laneM = lane & 15, kh = lane >> 4;
  int mrow = blockIdx.x * 64 + wave * 16;
  int arow = mrow + laneM;
  arow = arow < N_ ? arow : N_ - 1;

  v8f acc[8];
#pragma unroll
  for (int t = 0; t < 8; ++t) acc[t] = (v8f){0, 0, 0, 0, 0, 0, 0, 0};

  for (int r = 0; r < R_; ++r) {
    const float* arp = P + ((size_t)(b * R_ + r)) * N_ * C_ + (size_t)arow * C_;
    const float* wr = Wms + (size_t)r * C_ * C_;
#pragma unroll
    for (int c0 = 0; c0 < C_; c0 += 32) {
      V16u av;
      const float4* p0 = (const float4*)(arp + c0 + kh * 16);
      const float4* p1 = (const float4*)(arp + c0 + kh * 16 + 8);
      cvt8(av, 0, p0[0], p0[1]);
      cvt8(av, 8, p1[0], p1[1]);
      V16u bv[8];
#pragma unroll
      for (int t = 0; t < 8; ++t) {
        const float4* wp =
            (const float4*)(wr + (size_t)(t * 16 + laneM) * C_ + c0 + kh * 16);
        cvt8(bv[t], 0, wp[0], wp[1]);
        cvt8(bv[t], 8, wp[2], wp[3]);
      }
#pragma unroll
      for (int t = 0; t < 8; ++t) acc[t] = wmma_bf16(av.v, bv[t].v, acc[t]);
    }
  }

  float cb = conv_b[l];
#pragma unroll
  for (int t = 0; t < 8; ++t) {
#pragma unroll
    for (int vv = 0; vv < 8; ++vv) {
      int m = mrow + vv + 8 * kh;
      if (m < N_)
        hdiff[((size_t)b * N_ + m) * C_ + t * 16 + laneM] =
            fmaxf(acc[t][vv] + cb, 0.f);
    }
  }
}

// ------- generic WMMA GEMM: D = act(A0 @ W[:, :128]^T (+ A1 @ W[:,128:]^T) + biases)
__global__ __launch_bounds__(128) void gemm128_kernel(
    const float* __restrict__ A0, const float* __restrict__ A1,
    const float* __restrict__ W, int ldw, const float* __restrict__ bias0,
    const float* __restrict__ bias1, int relu, float* __restrict__ D, int M) {
  int wave = threadIdx.x >> 5, lane = threadIdx.x & 31;
  int laneM = lane & 15, kh = lane >> 4;
  int mrow = blockIdx.x * 64 + wave * 16;
  int arow = mrow + laneM;
  arow = arow < M ? arow : M - 1;

  v8f acc[8];
#pragma unroll
  for (int t = 0; t < 8; ++t) acc[t] = (v8f){0, 0, 0, 0, 0, 0, 0, 0};

  int nseg = A1 ? 2 : 1;
  for (int seg = 0; seg < nseg; ++seg) {
    const float* A = seg ? A1 : A0;
    const float* arp = A + (size_t)arow * C_;
    const float* wsg = W + seg * C_;
#pragma unroll
    for (int c0 = 0; c0 < C_; c0 += 32) {
      V16u av;
      const float4* p0 = (const float4*)(arp + c0 + kh * 16);
      const float4* p1 = (const float4*)(arp + c0 + kh * 16 + 8);
      cvt8(av, 0, p0[0], p0[1]);
      cvt8(av, 8, p1[0], p1[1]);
      V16u bv[8];
#pragma unroll
      for (int t = 0; t < 8; ++t) {
        const float4* wp =
            (const float4*)(wsg + (size_t)(t * 16 + laneM) * ldw + c0 + kh * 16);
        cvt8(bv[t], 0, wp[0], wp[1]);
        cvt8(bv[t], 8, wp[2], wp[3]);
      }
#pragma unroll
      for (int t = 0; t < 8; ++t) acc[t] = wmma_bf16(av.v, bv[t].v, acc[t]);
    }
  }

#pragma unroll
  for (int t = 0; t < 8; ++t) {
    int col = t * 16 + laneM;
    float bb = bias0[col] + (bias1 ? bias1[col] : 0.f);
#pragma unroll
    for (int vv = 0; vv < 8; ++vv) {
      int m = mrow + vv + 8 * kh;
      if (m < M) {
        float s = acc[t][vv] + bb;
        if (relu) s = fmaxf(s, 0.f);
        D[(size_t)m * C_ + col] = s;
      }
    }
  }
}

// -------- retention core (S=1): score_h = sum_d q*k;  opre = score * v
__global__ __launch_bounds__(128) void ret_kernel(const float* __restrict__ q,
                                                  const float* __restrict__ k,
                                                  const float* __restrict__ v,
                                                  float* __restrict__ opre) {
  size_t idx = (size_t)blockIdx.x * C_ + threadIdx.x;  // wave == head (HD==32)
  float score = waveSum(q[idx] * k[idx]);
  opre[idx] = score * v[idx];
}

// -------- GroupNorm per (node, head) over 32 channels (one wave)
__global__ __launch_bounds__(128) void gn_kernel(const float* __restrict__ o,
                                                 const float* __restrict__ gng,
                                                 const float* __restrict__ gnb,
                                                 int l, float* __restrict__ hr) {
  size_t idx = (size_t)blockIdx.x * C_ + threadIdx.x;
  int lc = l * C_ + threadIdx.x;
  float x = o[idx];
  float mu = waveSum(x) * (1.f / 32.f);
  float d = x - mu;
  float var = waveSum(d * d) * (1.f / 32.f);
  hr[idx] = d * __frsqrt_rn(var + EPS_) * gng[lc] + gnb[lc];
}

// -------- final 2-wide head
__global__ __launch_bounds__(64) void out2_kernel(const float* __restrict__ hmid,
                                                  const float* __restrict__ ow2,
                                                  const float* __restrict__ ob2,
                                                  float* __restrict__ out) {
  int node = blockIdx.x;
  int o = threadIdx.x >> 5, lane = threadIdx.x & 31;
  const float* hm = hmid + (size_t)node * C_;
  const float* w = ow2 + o * C_;
  float s = 0.f;
#pragma unroll
  for (int j = 0; j < C_; j += 32) s = fmaf(hm[j + lane], w[j + lane], s);
  s = waveSum(s);
  if (lane == 0) out[(size_t)node * OUT_ + o] = s + ob2[o];
}

extern "C" void kernel_launch(void* const* d_in, const int* in_sizes, int n_in,
                              void* d_out, int out_size, void* d_ws,
                              size_t ws_size, hipStream_t stream) {
  const float* x = (const float*)d_in[0];
  const float* adj = (const float*)d_in[1];
  const float* emb_w = (const float*)d_in[2];
  const float* emb_b = (const float*)d_in[3];
  const float* alpha = (const float*)d_in[4];
  const float* transition = (const float*)d_in[5];
  const float* conv_w = (const float*)d_in[6];
  const float* conv_b = (const float*)d_in[7];
  const float* qw = (const float*)d_in[8];
  const float* qb = (const float*)d_in[9];
  const float* kw = (const float*)d_in[10];
  const float* kb = (const float*)d_in[11];
  const float* vw = (const float*)d_in[12];
  const float* vb = (const float*)d_in[13];
  const float* ow = (const float*)d_in[14];
  const float* ob = (const float*)d_in[15];
  const float* gng = (const float*)d_in[16];
  const float* gnb = (const float*)d_in[17];
  const float* w1 = (const float*)d_in[18];
  const float* b1 = (const float*)d_in[19];
  const float* eb1 = (const float*)d_in[20];
  const float* w2 = (const float*)d_in[21];
  const float* b2 = (const float*)d_in[22];
  const float* eb2 = (const float*)d_in[23];
  const float* ow1 = (const float*)d_in[24];
  const float* ob1 = (const float*)d_in[25];
  const float* ow2 = (const float*)d_in[26];
  const float* ob2 = (const float*)d_in[27];
  float* out = (float*)d_out;

  float* ws = (float*)d_ws;
  const size_t HN = (size_t)M_ * C_;  // 384000
  float* hbuf[2] = {ws, ws + HN};
  float* hpbuf[2] = {ws + 2 * HN, ws + 3 * HN};
  float* P = ws + 4 * HN;                      // B*R*N*C
  float* Wms = P + (size_t)B_ * R_ * N_ * C_;  // R*C*C
  float* asum = Wms + (size_t)R_ * C_ * C_;
  float* hdiff = asum + 8;   // also reused as opre / hmid
  float* qbuf = hdiff + HN;  // also reused as obuf
  float* kbuf = qbuf + HN;   // also reused as hr
  float* vbuf = kbuf + HN;
  __bf16* abf = (__bf16*)(vbuf + HN);  // bf16 copy of adj (45 MB)

  const int nElem = (int)HN;
  const size_t adjElem = (size_t)B_ * R_ * N_ * N_;
  zero_kernel<<<dim3((nElem + 255) / 256), 256, 0, stream>>>(hpbuf[0], nElem);
  embed_kernel<<<dim3((nElem + 255) / 256), 256, 0, stream>>>(x, emb_w, emb_b,
                                                              hbuf[0]);
  cvt_adj_kernel<<<dim3((unsigned)((adjElem / 4 + 255) / 256)), 256, 0,
                   stream>>>(adj, abf);

  const int rbDiff = (N_ + 63) / 64;  // 24
  const int rbGemm = (M_ + 63) / 64;  // 47
  int cur = 0, curp = 0;
  for (int l = 0; l < L_; ++l) {
    size_t lCC = (size_t)l * C_ * C_;
    size_t lC = (size_t)l * C_;
    asum_kernel<<<1, 32, 0, stream>>>(alpha, l, asum);
    prep_kernel<<<dim3((R_ * C_ * C_ + 255) / 256), 256, 0, stream>>>(
        transition, conv_w, l, Wms);
    diffusion_kernel<<<dim3(rbDiff, B_ * R_), 128, 0, stream>>>(
        abf, hbuf[cur], asum, P);
    mix_kernel<<<dim3(rbDiff, B_), 128, 0, stream>>>(P, Wms, conv_b, l, hdiff);
    // q, k, v projections
    gemm128_kernel<<<dim3(rbGemm), 128, 0, stream>>>(
        hdiff, nullptr, qw + lCC, C_, qb + lC, nullptr, 0, qbuf, M_);
    gemm128_kernel<<<dim3(rbGemm), 128, 0, stream>>>(
        hdiff, nullptr, kw + lCC, C_, kb + lC, nullptr, 0, kbuf, M_);
    gemm128_kernel<<<dim3(rbGemm), 128, 0, stream>>>(
        hdiff, nullptr, vw + lCC, C_, vb + lC, nullptr, 0, vbuf, M_);
    // retention core (overwrites hdiff with o_pre)
    ret_kernel<<<dim3(M_), 128, 0, stream>>>(qbuf, kbuf, vbuf, hdiff);
    // o-projection (into qbuf)
    gemm128_kernel<<<dim3(rbGemm), 128, 0, stream>>>(
        hdiff, nullptr, ow + lCC, C_, ob + lC, nullptr, 0, qbuf, M_);
    // GroupNorm -> hr (into kbuf)
    gn_kernel<<<dim3(M_), 128, 0, stream>>>(qbuf, gng, gnb, l, kbuf);
    // h_prime_new = relu(hp_old @ w1^T + b1 + eb1)
    gemm128_kernel<<<dim3(rbGemm), 128, 0, stream>>>(
        hpbuf[curp], nullptr, w1 + lCC, C_, b1 + lC, eb1 + lC, 1,
        hpbuf[1 - curp], M_);
    // h_new = relu([hr | hp_old] @ w2^T + b2 + eb2)
    gemm128_kernel<<<dim3(rbGemm), 128, 0, stream>>>(
        kbuf, hpbuf[curp], w2 + (size_t)l * C_ * 2 * C_, 2 * C_, b2 + lC,
        eb2 + lC, 1, hbuf[1 - cur], M_);
    cur = 1 - cur;
    curp = 1 - curp;
  }
  // output head
  gemm128_kernel<<<dim3(rbGemm), 128, 0, stream>>>(hbuf[cur], nullptr, ow1, C_,
                                                   ob1, nullptr, 1, hdiff, M_);
  out2_kernel<<<dim3(M_), 64, 0, stream>>>(hdiff, ow2, ob2, out);
}
